// HNHN_7670811591238
// MI455X (gfx1250) — matrix operand order repaired and checked by reading, not compiled, not run on hardware
//
#include <hip/hip_runtime.h>
#include <hip/hip_bf16.h>
#include <stddef.h>

// ---------------------------------------------------------------------------
// HNHN hypergraph layer for MI455X (gfx1250, wave32, WMMA).
//   Wh_v       = vfeat @ (W1@Wve) + (b1@Wve + bve)        [fused 2 GEMMs]
//   feat_e     = segsum_e( (v_w[n]/e_s[e]) * Wh_v[n] )    [atomic scatter]
//   Wh_e       = feat_e @ Wev + bev                        [GEMM]
//   feat_v_out = segsum_v( (e_w[e]/v_s[n]) * Wh_e[e] )    [atomic scatter]
// GEMMs use v_wmma_f32_16x16x32_bf16 with a bf16 hi/lo split of both
// operands (3 WMMAs/tile) for near-fp32 accuracy at bf16 matrix-pipe rates.
// Scatter uses hardware global_atomic_add_f32 (inline asm, vaddr+offset
// form) into L2-resident destinations.
// ---------------------------------------------------------------------------

typedef __attribute__((ext_vector_type(16))) __bf16 v16bf;
typedef __attribute__((ext_vector_type(8)))  __bf16 v8bf;
typedef __attribute__((ext_vector_type(8)))  float  v8f;

#define D_DIM 128

// ---------------------------------------------------------------- zero output
__global__ void __launch_bounds__(256)
hnhn_zero_f4(float4* __restrict__ p, long n4)
{
    long i = (long)blockIdx.x * blockDim.x + threadIdx.x;
    if (i < n4) p[i] = make_float4(0.f, 0.f, 0.f, 0.f);
}

// ------------------------------------------------- weight prep: Wc = W1 @ Wve
// Writes WcT (n-major / transposed) split into bf16 hi/lo, and fused bias
// bc = b1 @ Wve + bve.  One thread per (k, n) of the 128x128 product.
__global__ void __launch_bounds__(256)
hnhn_prep_wc(const float* __restrict__ W1, const float* __restrict__ Wve,
             const float* __restrict__ b1, const float* __restrict__ bve,
             __bf16* __restrict__ WcT_hi, __bf16* __restrict__ WcT_lo,
             float* __restrict__ bc)
{
    const int idx = blockIdx.x * blockDim.x + threadIdx.x;
    const int k = idx >> 7;
    const int n = idx & 127;
    float s = 0.f;
    for (int j = 0; j < D_DIM; ++j)
        s = fmaf(W1[k * D_DIM + j], Wve[j * D_DIM + n], s);
    const __bf16 h = (__bf16)s;
    WcT_hi[n * D_DIM + k] = h;
    WcT_lo[n * D_DIM + k] = (__bf16)(s - (float)h);
    if (k == 0) {
        float t = bve[n];
        for (int j = 0; j < D_DIM; ++j)
            t = fmaf(b1[j], Wve[j * D_DIM + n], t);
        bc[n] = t;
    }
}

// ------------------------------------- weight prep: transpose + hi/lo split
__global__ void __launch_bounds__(256)
hnhn_prep_split_t(const float* __restrict__ W,
                  __bf16* __restrict__ hi, __bf16* __restrict__ lo)
{
    const int idx = blockIdx.x * blockDim.x + threadIdx.x;
    const int k = idx >> 7;
    const int n = idx & 127;
    const float s = W[k * D_DIM + n];
    const __bf16 h = (__bf16)s;
    hi[n * D_DIM + k] = h;
    lo[n * D_DIM + k] = (__bf16)(s - (float)h);
}

// ------------------------------------------------------------- WMMA GEMM
// D[M,128] = A[M,128] @ B[128,128] + bias, B pre-split/transposed to bf16.
// One wave per 16-row strip; 8 n-tiles held in 8 v8f accumulators.
__global__ void __launch_bounds__(256)
hnhn_gemm_bf16x2(const float*  __restrict__ A,
                 const __bf16* __restrict__ BT_hi,   // [n][k] n-major
                 const __bf16* __restrict__ BT_lo,
                 const float*  __restrict__ bias,
                 float*        __restrict__ D,
                 int mtiles)
{
    const int lane = threadIdx.x & 31;
    const int wave = blockIdx.x * (blockDim.x >> 5) + (threadIdx.x >> 5);
    if (wave >= mtiles) return;                    // wave-uniform: EXEC stays full

    const float* __restrict__ Arow = A + (size_t)wave * 16 * D_DIM;

    v8f acc[8];
    const v8f vzero = {0.f, 0.f, 0.f, 0.f, 0.f, 0.f, 0.f, 0.f};
#pragma unroll
    for (int t = 0; t < 8; ++t) acc[t] = vzero;

    // A operand layout (16-bit 16x32): lanes 0-15 hold K 0-7 & 16-23,
    // lanes 16-31 hold K 8-15 & 24-31, M = lane%16.
    const int m      = lane & 15;
    const int khalfA = (lane < 16) ? 0 : 8;
    // B operand layout (16-bit 32x16): lanes 0-15 hold K 0-7 & 16-23,
    // lanes 16-31 hold K 8-15 & 24-31, N = lane%16.
    const int kgB  = (lane < 16) ? 0 : 8;
    const int nloc = lane & 15;

#pragma unroll
    for (int ks = 0; ks < 4; ++ks) {
        const int kk = ks * 32;

        // ---- load A tile (fp32) and split into bf16 hi/lo ----
        const float* pa = Arow + m * D_DIM + kk + khalfA;
        float av[16];
        {
            const float4 x0 = *(const float4*)(pa);
            const float4 x1 = *(const float4*)(pa + 4);
            const float4 x2 = *(const float4*)(pa + 16);
            const float4 x3 = *(const float4*)(pa + 20);
            av[0]=x0.x;  av[1]=x0.y;  av[2]=x0.z;  av[3]=x0.w;
            av[4]=x1.x;  av[5]=x1.y;  av[6]=x1.z;  av[7]=x1.w;
            av[8]=x2.x;  av[9]=x2.y;  av[10]=x2.z; av[11]=x2.w;
            av[12]=x3.x; av[13]=x3.y; av[14]=x3.z; av[15]=x3.w;
        }
        v16bf ahi, alo;
#pragma unroll
        for (int e = 0; e < 16; ++e) {
            const __bf16 h = (__bf16)av[e];
            ahi[e] = h;
            alo[e] = (__bf16)(av[e] - (float)h);
        }

#pragma unroll
        for (int nt = 0; nt < 8; ++nt) {
            const __bf16* pbh = BT_hi + (size_t)(nt * 16 + nloc) * D_DIM + kk + kgB;
            const __bf16* pbl = BT_lo + (size_t)(nt * 16 + nloc) * D_DIM + kk + kgB;
            v16bf bhi, blo;
            {
                const v8bf h0 = *(const v8bf*)(pbh);
                const v8bf h1 = *(const v8bf*)(pbh + 16);
                const v8bf l0 = *(const v8bf*)(pbl);
                const v8bf l1 = *(const v8bf*)(pbl + 16);
#pragma unroll
                for (int e = 0; e < 8; ++e) {
                    bhi[e] = h0[e]; bhi[8 + e] = h1[e];
                    blo[e] = l0[e]; blo[8 + e] = l1[e];
                }
            }
            // D = Ahi*Bhi + Ahi*Blo + Alo*Bhi + C   (~fp32 accuracy)
            acc[nt] = __builtin_amdgcn_wmma_f32_16x16x32_bf16(
                          false, ahi, false, bhi, (short)0, acc[nt], false, false);
            acc[nt] = __builtin_amdgcn_wmma_f32_16x16x32_bf16(
                          false, ahi, false, blo, (short)0, acc[nt], false, false);
            acc[nt] = __builtin_amdgcn_wmma_f32_16x16x32_bf16(
                          false, alo, false, bhi, (short)0, acc[nt], false, false);
        }
    }

    float bvals[8];
#pragma unroll
    for (int nt = 0; nt < 8; ++nt) bvals[nt] = bias[nt * 16 + nloc];

    // C/D layout: VGPR r holds row M = r (lanes 0-15) or 8+r (lanes 16-31).
    const int row0 = wave * 16 + ((lane < 16) ? 0 : 8);
#pragma unroll
    for (int r = 0; r < 8; ++r) {
        float* __restrict__ drow = D + (size_t)(row0 + r) * D_DIM + nloc;
#pragma unroll
        for (int nt = 0; nt < 8; ++nt)
            drow[nt * 16] = acc[nt][r] + bvals[nt];
    }
}

// ------------------------------------------------- gather/scale/atomic scatter
// out[dst[i]] += (w_num[src[i]] / w_den[dst[i]]) * rows[src[i]]
// One wave per nnz: float4 gather per lane (512B/wave), then 4 hardware
// global_atomic_add_f32 per lane off one vaddr pair (offset: field).
// Indices are wave-uniform -> readfirstlane so the weight loads and base
// address math scalarize.  Destinations are L2-resident (20MB / 102MB).
__global__ void __launch_bounds__(256)
hnhn_scatter_atomic(const float* __restrict__ rows,
                    const int*   __restrict__ src_idx,
                    const int*   __restrict__ dst_idx,
                    const float* __restrict__ w_num,
                    const float* __restrict__ w_den,
                    float*       __restrict__ out,
                    int nnz)
{
    const int lane = threadIdx.x & 31;
    const int i = blockIdx.x * (blockDim.x >> 5) + (threadIdx.x >> 5);
    if (i >= nnz) return;
    const int s = __builtin_amdgcn_readfirstlane(src_idx[i]);
    const int d = __builtin_amdgcn_readfirstlane(dst_idx[i]);
    const float w = w_num[s] / w_den[d];
    const float4 x = *(const float4*)(rows + (size_t)s * D_DIM + lane * 4);
    float* o = out + (size_t)d * D_DIM + lane * 4;
    // atomics-without-return (TH[0]=0): tracked by STOREcnt, drained at endpgm
    asm volatile("global_atomic_add_f32 %0, %1, off"           :: "v"(o), "v"(w * x.x) : "memory");
    asm volatile("global_atomic_add_f32 %0, %1, off offset:4"  :: "v"(o), "v"(w * x.y) : "memory");
    asm volatile("global_atomic_add_f32 %0, %1, off offset:8"  :: "v"(o), "v"(w * x.z) : "memory");
    asm volatile("global_atomic_add_f32 %0, %1, off offset:12" :: "v"(o), "v"(w * x.w) : "memory");
}

// ---------------------------------------------------------------------------
extern "C" void kernel_launch(void* const* d_in, const int* in_sizes, int n_in,
                              void* d_out, int out_size, void* d_ws, size_t ws_size,
                              hipStream_t stream)
{
    (void)n_in; (void)ws_size;

    const float* vfeat        = (const float*)d_in[0];
    // d_in[1] (efeat) unused by the reference math
    const float* v_reg_weight = (const float*)d_in[2];
    const float* v_reg_sum    = (const float*)d_in[3];
    const float* e_reg_weight = (const float*)d_in[4];
    const float* e_reg_sum    = (const float*)d_in[5];
    const int*   node_idx     = (const int*)d_in[6];
    const int*   edge_idx     = (const int*)d_in[7];
    const float* W1           = (const float*)d_in[8];
    const float* b1           = (const float*)d_in[9];
    const float* Wve          = (const float*)d_in[10];
    const float* bve          = (const float*)d_in[11];
    const float* Wev          = (const float*)d_in[12];
    const float* bev          = (const float*)d_in[13];

    const int n_v = in_sizes[2];   // 200000
    const int n_e = in_sizes[4];   // 40000
    const int nnz = in_sizes[6];   // 1000000

    float* feat_v_out = (float*)d_out;                          // [n_v,128]
    float* feat_e     = (float*)d_out + (size_t)n_v * D_DIM;    // [n_e,128]

    // workspace layout (Wh buffer reused for Wh_v then Wh_e)
    char* ws = (char*)d_ws;
    size_t off = 0;
    float*  Wh      = (float*)(ws + off);  off += (size_t)n_v * D_DIM * sizeof(float);
    __bf16* WcT_hi  = (__bf16*)(ws + off); off += (size_t)D_DIM * D_DIM * sizeof(__bf16);
    __bf16* WcT_lo  = (__bf16*)(ws + off); off += (size_t)D_DIM * D_DIM * sizeof(__bf16);
    __bf16* WevT_hi = (__bf16*)(ws + off); off += (size_t)D_DIM * D_DIM * sizeof(__bf16);
    __bf16* WevT_lo = (__bf16*)(ws + off); off += (size_t)D_DIM * D_DIM * sizeof(__bf16);
    float*  bc      = (float*)(ws + off);  off += (size_t)D_DIM * sizeof(float);

    // 1. zero both outputs (segment sums accumulate via atomics)
    {
        long n4 = (long)out_size / 4;
        int blocks = (int)((n4 + 255) / 256);
        hnhn_zero_f4<<<blocks, 256, 0, stream>>>((float4*)d_out, n4);
    }
    // 2. weight prep: Wc = W1@Wve (+fused bias), and Wev, both split/transposed
    hnhn_prep_wc<<<(D_DIM * D_DIM) / 256, 256, 0, stream>>>(
        W1, Wve, b1, bve, WcT_hi, WcT_lo, bc);
    hnhn_prep_split_t<<<(D_DIM * D_DIM) / 256, 256, 0, stream>>>(
        Wev, WevT_hi, WevT_lo);
    // 3. Wh_v = vfeat @ Wc + bc
    {
        int mt = n_v / 16;
        hnhn_gemm_bf16x2<<<(mt + 7) / 8, 256, 0, stream>>>(
            vfeat, WcT_hi, WcT_lo, bc, Wh, mt);
    }
    // 4. feat_e = segment_sum_e( (v_w/e_s) * Wh_v[node] )
    hnhn_scatter_atomic<<<(nnz + 7) / 8, 256, 0, stream>>>(
        Wh, node_idx, edge_idx, v_reg_weight, e_reg_sum, feat_e, nnz);
    // 5. Wh_e = feat_e @ Wev + bev   (reuses Wh buffer)
    {
        int mt = n_e / 16;
        hnhn_gemm_bf16x2<<<(mt + 7) / 8, 256, 0, stream>>>(
            feat_e, WevT_hi, WevT_lo, bev, Wh, mt);
    }
    // 6. feat_v_out = segment_sum_v( (e_w/v_s) * Wh_e[edge] )
    hnhn_scatter_atomic<<<(nnz + 7) / 8, 256, 0, stream>>>(
        Wh, edge_idx, node_idx, e_reg_weight, v_reg_sum, feat_v_out, nnz);
}
